// Bottleneck_3865470566931
// MI455X (gfx1250) — matrix operand order repaired and checked by reading, not compiled, not run on hardware
//
#include <hip/hip_runtime.h>
#include <hip/hip_bf16.h>

// ---------------------------------------------------------------------------
// Types for CDNA5 WMMA (wave32): v_wmma_f32_16x16x32_bf16
// ---------------------------------------------------------------------------
typedef __attribute__((ext_vector_type(16))) __bf16 bf16x16;
typedef __attribute__((ext_vector_type(8)))  __bf16 bf16x8;
typedef __attribute__((ext_vector_type(8)))  float  f32x8;

#define WMMA_BF16(a, b, c) \
    __builtin_amdgcn_wmma_f32_16x16x32_bf16(false, (a), false, (b), (short)0, (c), false, false)

static __device__ __forceinline__ bf16x16 join8(bf16x8 lo, bf16x8 hi) {
    return __builtin_shufflevector(lo, hi, 0,1,2,3,4,5,6,7,8,9,10,11,12,13,14,15);
}

static __device__ __forceinline__ bf16x8 cvt8(f32x8 v) {
    bf16x8 r;
#pragma unroll
    for (int i = 0; i < 8; ++i) r[i] = (__bf16)v[i];
    return r;
}

static __device__ __forceinline__ float gelu_exact(float v) {
    return 0.5f * v * (1.0f + erff(v * 0.70710678118654752440f));
}

// ---------------------------------------------------------------------------
// Problem constants
// ---------------------------------------------------------------------------
#define NPTS   200000
#define C_IN   256
#define C_MID  64
#define LN_EPS 1e-6f

// ---------------------------------------------------------------------------
// Kernel 0: weight transpose + f32->bf16 convert; zero the sentinel row.
// Layouts (B-fragment friendly: [out_col][k], k contiguous):
//   W1t [64][256], W2t [9][64][64] (= W2[k][c][d] -> W2t[k][d][c]), W3t [256][64]
// ---------------------------------------------------------------------------
__global__ void prep_kernel(const float* __restrict__ W1,
                            const float* __restrict__ W2,
                            const float* __restrict__ W3,
                            __bf16* __restrict__ W1t,
                            __bf16* __restrict__ W2t,
                            __bf16* __restrict__ W3t,
                            __bf16* __restrict__ h1_sentinel) {
    int i = blockIdx.x * 256 + threadIdx.x;
    if (i < 16384) {                               // W1t: n = i>>8, k = i&255
        int n = i >> 8, k = i & 255;
        W1t[i] = (__bf16)W1[k * 64 + n];
    } else if (i < 16384 + 36864) {                // W2t: kk,d,c
        int j = i - 16384;
        int kk = j >> 12, d = (j >> 6) & 63, c = j & 63;
        W2t[j] = (__bf16)W2[kk * 4096 + c * 64 + d];
    } else if (i < 16384 + 36864 + 16384) {        // W3t: n = j>>6, k = j&63
        int j = i - (16384 + 36864);
        int n = j >> 6, k = j & 63;
        W3t[j] = (__bf16)W3[k * 256 + n];
    } else if (i < 16384 + 36864 + 16384 + 64) {   // sentinel row of h1 = zeros
        h1_sentinel[i - (16384 + 36864 + 16384)] = (__bf16)0.0f;
    }
}

// ---------------------------------------------------------------------------
// Kernel 1: h1 = gelu(ln(x @ W1)) as bf16.  64 rows/block, 8 waves,
// wave tiling: rowTile = w>>1, two 16x16 col tiles. K=256 -> 8 wmma steps.
// ---------------------------------------------------------------------------
__global__ __launch_bounds__(256) void conv1_kernel(
    const float* __restrict__ x, const __bf16* __restrict__ W1t,
    const float* __restrict__ g1, const float* __restrict__ b1,
    __bf16* __restrict__ h1) {
    __shared__ float lds[64 * 66];
    __shared__ float smu[64], srs[64];

    const int tid   = threadIdx.x;
    const int lane  = tid & 31;
    const int w     = tid >> 5;
    const int rowBase = blockIdx.x * 64;
    const int rowTile = w >> 1;
    const int ct0     = (w & 1) * 2;       // two adjacent 16-col tiles
    const int m     = lane & 15;
    const int khalf = lane >> 4;

    f32x8 c0 = {}, c1 = {};
    const float*  arow = x + (size_t)(rowBase + rowTile * 16 + m) * C_IN;
    const __bf16* b0p  = W1t + (ct0 * 16 + m) * 256 + khalf * 16;
    const __bf16* b1p  = b0p + 16 * 256;

#pragma unroll
    for (int kc = 0; kc < 8; ++kc) {
        const int kb = kc * 32 + khalf * 16;
        f32x8 lo = *(const f32x8*)(arow + kb);
        f32x8 hi = *(const f32x8*)(arow + kb + 8);
        bf16x16 a  = join8(cvt8(lo), cvt8(hi));
        bf16x16 bb0 = *(const bf16x16*)(b0p + kc * 32);
        bf16x16 bb1 = *(const bf16x16*)(b1p + kc * 32);
        c0 = WMMA_BF16(a, bb0, c0);
        c1 = WMMA_BF16(a, bb1, c1);
    }

    const int r0 = rowTile * 16 + khalf * 8;
#pragma unroll
    for (int r = 0; r < 8; ++r) {
        lds[(r0 + r) * 66 + ct0 * 16 + m]       = c0[r];
        lds[(r0 + r) * 66 + (ct0 + 1) * 16 + m] = c1[r];
    }
    __syncthreads();

    if (tid < 64) {
        float s = 0.f, s2 = 0.f;
#pragma unroll
        for (int j = 0; j < 64; ++j) { float v = lds[tid * 66 + j]; s += v; s2 += v * v; }
        float mu  = s * (1.0f / 64.0f);
        float var = s2 * (1.0f / 64.0f) - mu * mu;
        smu[tid] = mu;
        srs[tid] = rsqrtf(var + LN_EPS);
    }
    __syncthreads();

#pragma unroll
    for (int i = 0; i < 16; ++i) {
        int idx = i * 256 + tid;
        int row = idx >> 6, col = idx & 63;
        float v = lds[row * 66 + col];
        float y = (v - smu[row]) * srs[row] * g1[col] + b1[col];
        h1[(size_t)(rowBase + row) * 64 + col] = (__bf16)gelu_exact(y);
    }
}

// ---------------------------------------------------------------------------
// Kernel 2: conv2 gather-GEMM (9 kernel maps, sentinel row = zeros) + LN + GELU.
// Same 64-row / 8-wave tiling. K=64 per map -> 2 wmma steps, 9 maps accumulated.
// h1 (25.6 MB) is fully L2-resident, so the gather hits L2.
// ---------------------------------------------------------------------------
__global__ __launch_bounds__(256) void conv2_kernel(
    const __bf16* __restrict__ h1, const __bf16* __restrict__ W2t,
    const int* __restrict__ nbr,
    const float* __restrict__ g2, const float* __restrict__ b2,
    __bf16* __restrict__ h2) {
    __shared__ float lds[64 * 66];
    __shared__ float smu[64], srs[64];

    const int tid   = threadIdx.x;
    const int lane  = tid & 31;
    const int w     = tid >> 5;
    const int rowBase = blockIdx.x * 64;
    const int rowTile = w >> 1;
    const int ct0     = (w & 1) * 2;
    const int m     = lane & 15;
    const int khalf = lane >> 4;

    f32x8 c0 = {}, c1 = {};
    const int gRow = rowBase + rowTile * 16 + m;

#pragma unroll
    for (int kk = 0; kk < 9; ++kk) {
        const int ridx = nbr[kk * NPTS + gRow];                 // sentinel NPTS -> zero row
        const __bf16* arow = h1 + (size_t)ridx * 64;
        const __bf16* wk   = W2t + kk * 4096 + (ct0 * 16 + m) * 64 + khalf * 16;
#pragma unroll
        for (int kc = 0; kc < 2; ++kc) {
            const int kb = kc * 32 + khalf * 16;
            bf16x16 a = join8(*(const bf16x8*)(arow + kb),
                              *(const bf16x8*)(arow + kb + 8));
            bf16x16 bb0 = *(const bf16x16*)(wk + kc * 32);
            bf16x16 bb1 = *(const bf16x16*)(wk + 16 * 64 + kc * 32);
            c0 = WMMA_BF16(a, bb0, c0);
            c1 = WMMA_BF16(a, bb1, c1);
        }
    }

    const int r0 = rowTile * 16 + khalf * 8;
#pragma unroll
    for (int r = 0; r < 8; ++r) {
        lds[(r0 + r) * 66 + ct0 * 16 + m]       = c0[r];
        lds[(r0 + r) * 66 + (ct0 + 1) * 16 + m] = c1[r];
    }
    __syncthreads();

    if (tid < 64) {
        float s = 0.f, s2 = 0.f;
#pragma unroll
        for (int j = 0; j < 64; ++j) { float v = lds[tid * 66 + j]; s += v; s2 += v * v; }
        float mu  = s * (1.0f / 64.0f);
        float var = s2 * (1.0f / 64.0f) - mu * mu;
        smu[tid] = mu;
        srs[tid] = rsqrtf(var + LN_EPS);
    }
    __syncthreads();

#pragma unroll
    for (int i = 0; i < 16; ++i) {
        int idx = i * 256 + tid;
        int row = idx >> 6, col = idx & 63;
        float v = lds[row * 66 + col];
        float y = (v - smu[row]) * srs[row] * g2[col] + b2[col];
        h2[(size_t)(rowBase + row) * 64 + col] = (__bf16)gelu_exact(y);
    }
}

// ---------------------------------------------------------------------------
// Kernel 3: out = gelu(ln(h2 @ W3) + x).  32 rows/block, 8 waves:
// rowTile = w>>2, 4 col tiles each (2x16 tile grid over 32x256 output).
// A fragments (K=64) loaded once, reused across all col tiles.
// ---------------------------------------------------------------------------
__global__ __launch_bounds__(256) void conv3_kernel(
    const __bf16* __restrict__ h2, const __bf16* __restrict__ W3t,
    const float* __restrict__ x,
    const float* __restrict__ g3, const float* __restrict__ b3,
    float* __restrict__ out) {
    __shared__ float lds[32 * 258];
    __shared__ float smu[32], srs[32];

    const int tid   = threadIdx.x;
    const int lane  = tid & 31;
    const int w     = tid >> 5;
    const int rowBase = blockIdx.x * 32;
    const int rowTile = w >> 2;          // 0..1
    const int ctb     = (w & 3) * 4;     // 0,4,8,12
    const int m     = lane & 15;
    const int khalf = lane >> 4;

    bf16x16 a0, a1;
    {
        const __bf16* arow = h2 + (size_t)(rowBase + rowTile * 16 + m) * 64;
        const int kb = khalf * 16;
        a0 = join8(*(const bf16x8*)(arow + kb),      *(const bf16x8*)(arow + kb + 8));
        a1 = join8(*(const bf16x8*)(arow + 32 + kb), *(const bf16x8*)(arow + 32 + kb + 8));
    }

    f32x8 c[4] = {};
#pragma unroll
    for (int t = 0; t < 4; ++t) {
        const __bf16* wp = W3t + ((ctb + t) * 16 + m) * 64 + khalf * 16;
        bf16x16 b0 = *(const bf16x16*)(wp);
        bf16x16 b1 = *(const bf16x16*)(wp + 32);
        c[t] = WMMA_BF16(a0, b0, c[t]);
        c[t] = WMMA_BF16(a1, b1, c[t]);
    }

    const int r0 = rowTile * 16 + khalf * 8;
#pragma unroll
    for (int t = 0; t < 4; ++t)
#pragma unroll
        for (int r = 0; r < 8; ++r)
            lds[(r0 + r) * 258 + (ctb + t) * 16 + m] = c[t][r];
    __syncthreads();

    if (tid < 32) {
        float s = 0.f, s2 = 0.f;
        for (int j = 0; j < 256; ++j) { float v = lds[tid * 258 + j]; s += v; s2 += v * v; }
        float mu  = s * (1.0f / 256.0f);
        float var = s2 * (1.0f / 256.0f) - mu * mu;
        smu[tid] = mu;
        srs[tid] = rsqrtf(var + LN_EPS);
    }
    __syncthreads();

    const float gv = g3[tid];
    const float bv = b3[tid];
#pragma unroll 4
    for (int r = 0; r < 32; ++r) {
        float v = lds[r * 258 + tid];
        float y = (v - smu[r]) * srs[r] * gv + bv;
        y += x[(size_t)(rowBase + r) * C_IN + tid];
        out[(size_t)(rowBase + r) * C_IN + tid] = gelu_exact(y);
    }
}

// ---------------------------------------------------------------------------
// Launch: prep -> conv1 -> conv2 -> conv3 (all on `stream`).
// Workspace: W1t(32KB) | W2t(72KB) | W3t(32KB) | h1 bf16 (N+1)x64 | h2 bf16 Nx64
// ---------------------------------------------------------------------------
extern "C" void kernel_launch(void* const* d_in, const int* in_sizes, int n_in,
                              void* d_out, int out_size, void* d_ws, size_t ws_size,
                              hipStream_t stream) {
    (void)in_sizes; (void)n_in; (void)out_size; (void)ws_size;

    const float* x  = (const float*)d_in[0];
    const float* W1 = (const float*)d_in[1];
    const float* W2 = (const float*)d_in[2];
    const float* W3 = (const float*)d_in[3];
    const float* g1 = (const float*)d_in[4];
    const float* b1 = (const float*)d_in[5];
    const float* g2 = (const float*)d_in[6];
    const float* b2 = (const float*)d_in[7];
    const float* g3 = (const float*)d_in[8];
    const float* b3 = (const float*)d_in[9];
    const int*  nbr = (const int*)d_in[10];
    float* out = (float*)d_out;

    char* ws = (char*)d_ws;
    __bf16* W1t = (__bf16*)(ws);                                   // 32768 B
    __bf16* W2t = (__bf16*)(ws + 32768);                           // 73728 B
    __bf16* W3t = (__bf16*)(ws + 32768 + 73728);                   // 32768 B
    __bf16* h1  = (__bf16*)(ws + 139264);                          // (N+1)*64*2 B
    __bf16* h2  = (__bf16*)(ws + 139264 + (size_t)(NPTS + 1) * 64 * 2);

    prep_kernel<<<(16384 + 36864 + 16384 + 64 + 255) / 256, 256, 0, stream>>>(
        W1, W2, W3, W1t, W2t, W3t, h1 + (size_t)NPTS * 64);
    conv1_kernel<<<NPTS / 64, 256, 0, stream>>>(x, W1t, g1, b1, h1);
    conv2_kernel<<<NPTS / 64, 256, 0, stream>>>(h1, W2t, nbr, g2, b2, h2);
    conv3_kernel<<<NPTS / 32, 256, 0, stream>>>(h2, W3t, x, g3, b3, out);
}
